// MixtralDecoderLayer_16587163697426
// MI455X (gfx1250) — matrix operand order repaired and checked by reading, not compiled, not run on hardware
//
#include <hip/hip_runtime.h>
#include <hip/hip_bf16.h>
#include <math.h>
#include <stdint.h>

// ---------------------------------------------------------------------------
// Mixtral decoder layer for gfx1250 (MI455X), bf16 WMMA everywhere.
// B=1, S=2048, H=1024, NH=16, KVH=4, HD=64, I=3584, E=8, TOPK=2.
// A-tiles: async global->LDS double-buffering (ASYNCcnt) in QKV/O-proj/FFN2,
// whole-tile Tensor Data Mover load (TENSORcnt) with LDS padding in FFN1.
// ---------------------------------------------------------------------------

typedef __bf16 bf16_t;
typedef __attribute__((ext_vector_type(16))) __bf16 v16bf;
typedef __attribute__((ext_vector_type(8)))  __bf16 v8bf;
typedef __attribute__((ext_vector_type(8)))  float  v8f;
typedef __attribute__((ext_vector_type(4)))  unsigned v4u;
typedef __attribute__((ext_vector_type(8)))  unsigned v8u;

#define S_LEN   2048
#define HDIM    1024
#define NHEADS  16
#define KVHEADS 4
#define HEADD   64
#define IDIM    3584
#define NEXP    8

// LDS A-tile (async path): 64 rows x 32 bf16, padded row stride 40 elems (80B):
// 16B-aligned for b128, and row*20 mod 64 banks distinct across 16 rows.
#define ATILE_STRIDE 40

// TDM-staged tile (ffn1): 64 rows x 1024 bf16, 16B pad per 1024B of data
// => padded row stride 2080B (= 130*16B, banks row*8 mod 64).
#define TDM_ROW_BYTES 2080

__device__ __forceinline__ v8f wmma_bf16(v16bf a, v16bf b, v8f c) {
  return __builtin_amdgcn_wmma_f32_16x16x32_bf16(false, a, false, b, (short)0, c,
                                                 false, false);
}

// LDS byte offset of a __shared__ object (AS3 pointers are 32-bit LDS offsets).
__device__ __forceinline__ unsigned lds_off_of(const void* p) {
  return (unsigned)(unsigned long long)
      (__attribute__((address_space(3))) const void*)p;
}

// Stage 32 contiguous bytes from global to LDS asynchronously:
// two b128 async transfers, tracked by ASYNCcnt.
__device__ __forceinline__ void async_stage_rowpair(unsigned lds_byte_off,
                                                    const bf16_t* gp) {
  asm volatile(
      "global_load_async_to_lds_b128 %0, %1, off\n\t"
      "global_load_async_to_lds_b128 %0, %1, off offset:16"
      :
      : "v"(lds_byte_off), "v"((unsigned long long)(size_t)gp)
      : "memory");
}
#define S_WAIT_ASYNCCNT(n) asm volatile("s_wait_asynccnt " #n ::: "memory")

__device__ __forceinline__ void wait_tensorcnt0() {
#if __has_builtin(__builtin_amdgcn_s_wait_tensorcnt)
  __builtin_amdgcn_s_wait_tensorcnt(0);
#else
  asm volatile("s_wait_tensorcnt 0x0" ::: "memory");
#endif
}

// A fragment: 16x32 bf16 tile from row-major A (lda in elements).
// lane<16 : row=lane,   K = k0+0..7   and k0+16..23
// lane>=16: row=lane-16,K = k0+8..15  and k0+24..31   (per ISA 7.12.2)
__device__ __forceinline__ v16bf load_a_bf16(const bf16_t* A, int lda, int m0,
                                             int k0, int lane) {
  const bf16_t* p = A + (size_t)(m0 + (lane & 15)) * lda + k0 + ((lane >> 4) << 3);
  v8bf lo = *(const v8bf*)p;
  v8bf hi = *(const v8bf*)(p + 16);
  return __builtin_shufflevector(lo, hi, 0, 1, 2, 3, 4, 5, 6, 7, 8, 9, 10, 11,
                                 12, 13, 14, 15);
}

// A fragment from the TDM-padded LDS tile (16B pad every 512 elements).
__device__ __forceinline__ v16bf load_a_tdm(const bf16_t* base, int r0, int k,
                                            int lane) {
  int row = r0 + (lane & 15);
  int kk = k + ((lane >> 4) << 3);
  const char* p = (const char*)base + row * TDM_ROW_BYTES + kk * 2 + ((kk >> 9) << 4);
  v8bf lo = *(const v8bf*)p;
  v8bf hi = *(const v8bf*)(p + 32);
  return __builtin_shufflevector(lo, hi, 0, 1, 2, 3, 4, 5, 6, 7, 8, 9, 10, 11,
                                 12, 13, 14, 15);
}

// B fragment from fp32 weight W[N][K] (contraction along K columns).
// n = n0+(lane&15); element i = k0+8*(lane>>4)+i (i<8), +16 for i>=8.
__device__ __forceinline__ v16bf load_b_w32(const float* W, int ldw, int n0,
                                            int k0, int lane) {
  const float* p = W + (size_t)(n0 + (lane & 15)) * ldw + k0 + ((lane >> 4) << 3);
  v16bf b;
#pragma unroll
  for (int i = 0; i < 16; ++i) b[i] = (bf16_t)p[i < 8 ? i : i + 8];
  return b;
}

// B fragment from bf16 matrix M[N][K] (contraction along K columns).
__device__ __forceinline__ v16bf load_b_bf16_col(const bf16_t* M, int ldm, int n0,
                                                 int k0, int lane) {
  const bf16_t* p = M + (size_t)(n0 + (lane & 15)) * ldm + k0 + ((lane >> 4) << 3);
  v16bf b;
#pragma unroll
  for (int i = 0; i < 16; ++i) b[i] = p[i < 8 ? i : i + 8];
  return b;
}

// B fragment from bf16 matrix M[K][N] (contraction along rows): strided over k.
__device__ __forceinline__ v16bf load_b_bf16_row(const bf16_t* M, int ldm, int k0,
                                                 int n0, int lane) {
  const bf16_t* p = M + (size_t)(k0 + ((lane >> 4) << 3)) * ldm + n0 + (lane & 15);
  v16bf b;
#pragma unroll
  for (int i = 0; i < 16; ++i) b[i] = p[(size_t)(i < 8 ? i : i + 8) * ldm];
  return b;
}

// ---------------------------------------------------------------------------
// RMSNorm: fp32 in -> bf16 out. One block (8 waves) per token.
// ---------------------------------------------------------------------------
__global__ void rmsnorm_kernel(const float* __restrict__ x,
                               const float* __restrict__ w,
                               bf16_t* __restrict__ out) {
  int t = blockIdx.x;
  const float* row = x + (size_t)t * HDIM;
  float s = 0.f;
  for (int i = threadIdx.x; i < HDIM; i += blockDim.x) {
    float v = row[i];
    s += v * v;
  }
#pragma unroll
  for (int m = 16; m >= 1; m >>= 1) s += __shfl_xor(s, m, 32);
  __shared__ float red[8];
  int wid = threadIdx.x >> 5, lane = threadIdx.x & 31;
  if (lane == 0) red[wid] = s;
  __syncthreads();
  float tot = 0.f;
#pragma unroll
  for (int i = 0; i < 8; ++i) tot += red[i];
  float inv = rsqrtf(tot / (float)HDIM + 1e-6f);
  for (int i = threadIdx.x; i < HDIM; i += blockDim.x)
    out[(size_t)t * HDIM + i] = (bf16_t)(row[i] * inv * w[i]);
}

// ---------------------------------------------------------------------------
// Fused QKV projection + RoPE. grid=(S/64, 24): g<16 Q-head, <20 K-head, else V.
// A tiles double-buffered via async global->LDS.
// ---------------------------------------------------------------------------
__global__ void qkv_rope_kernel(const bf16_t* __restrict__ xb,
                                const float* __restrict__ wq,
                                const float* __restrict__ wk,
                                const float* __restrict__ wv,
                                const float* __restrict__ cosb,
                                const float* __restrict__ sinb,
                                bf16_t* __restrict__ qb, bf16_t* __restrict__ kb,
                                bf16_t* __restrict__ vb) {
  __shared__ bf16_t atile[2][64 * ATILE_STRIDE];
  int tid = threadIdx.x;
  int lane = tid & 31, wid = tid >> 5;
  int row = tid >> 1, seg = tid & 1;
  int mblk = blockIdx.x * 64;
  int m0 = mblk + wid * 16;
  int g = blockIdx.y;
  const float* W;
  bf16_t* outp;
  int n0;
  bool rope;
  if (g < 16) {
    W = wq; outp = qb + (size_t)g * S_LEN * HEADD; n0 = g * 64; rope = true;
  } else if (g < 20) {
    W = wk; outp = kb + (size_t)(g - 16) * S_LEN * HEADD; n0 = (g - 16) * 64; rope = true;
  } else {
    W = wv; outp = vb + (size_t)(g - 20) * S_LEN * HEADD; n0 = (g - 20) * 64; rope = false;
  }
  const bf16_t* grow = xb + (size_t)(mblk + row) * HDIM + seg * 16;
  unsigned lrow0 = lds_off_of(&atile[0][0]) + (unsigned)(row * 80 + seg * 32);
  unsigned lrow1 = lds_off_of(&atile[1][0]) + (unsigned)(row * 80 + seg * 32);

  v8f acc[4] = {};
  async_stage_rowpair(lrow0, grow);
  for (int k = 0; k < HDIM; k += 32) {
    int cur = (k >> 5) & 1;
    if (k + 32 < HDIM) {
      async_stage_rowpair(cur ? lrow0 : lrow1, grow + (k + 32));
      S_WAIT_ASYNCCNT(2);
    } else {
      S_WAIT_ASYNCCNT(0);
    }
    v16bf a = load_a_bf16(atile[cur], ATILE_STRIDE, wid * 16, 0, lane);
#pragma unroll
    for (int j = 0; j < 4; ++j)
      acc[j] = wmma_bf16(a, load_b_w32(W, HDIM, n0 + 16 * j, k, lane), acc[j]);
  }
  int half = lane >> 4, col = lane & 15;
#pragma unroll
  for (int r = 0; r < 8; ++r) {
    int srow = m0 + r + 8 * half;
    if (rope) {
      float v0 = acc[0][r], v1 = acc[1][r], v2 = acc[2][r], v3 = acc[3][r];
      float c0 = cosb[srow * 64 + col],      s0 = sinb[srow * 64 + col];
      float c1 = cosb[srow * 64 + 16 + col], s1 = sinb[srow * 64 + 16 + col];
      float c2 = cosb[srow * 64 + 32 + col], s2 = sinb[srow * 64 + 32 + col];
      float c3 = cosb[srow * 64 + 48 + col], s3 = sinb[srow * 64 + 48 + col];
      outp[(size_t)srow * 64 + col]      = (bf16_t)(v0 * c0 - v2 * s0);
      outp[(size_t)srow * 64 + 16 + col] = (bf16_t)(v1 * c1 - v3 * s1);
      outp[(size_t)srow * 64 + 32 + col] = (bf16_t)(v2 * c2 + v0 * s2);
      outp[(size_t)srow * 64 + 48 + col] = (bf16_t)(v3 * c3 + v1 * s3);
    } else {
#pragma unroll
      for (int j = 0; j < 4; ++j)
        outp[(size_t)srow * 64 + 16 * j + col] = (bf16_t)acc[j][r];
    }
  }
}

// ---------------------------------------------------------------------------
// Causal flash attention (GQA 16 Q heads over 4 KV heads).
// ---------------------------------------------------------------------------
__global__ void attn_kernel(const bf16_t* __restrict__ Q,
                            const bf16_t* __restrict__ K,
                            const bf16_t* __restrict__ V,
                            bf16_t* __restrict__ O) {
  int lane = threadIdx.x & 31;
  int head = blockIdx.y;
  int m0 = blockIdx.x * 16;
  const bf16_t* Qh = Q + (size_t)head * S_LEN * HEADD;
  const bf16_t* Kh = K + (size_t)(head >> 2) * S_LEN * HEADD;
  const bf16_t* Vh = V + (size_t)(head >> 2) * S_LEN * HEADD;
  __shared__ bf16_t pbuf[16 * 40];

  v16bf qa0 = load_a_bf16(Qh, HEADD, m0, 0, lane);
  v16bf qa1 = load_a_bf16(Qh, HEADD, m0, 32, lane);

  float rmax[8], rsum[8];
#pragma unroll
  for (int r = 0; r < 8; ++r) { rmax[r] = -1e30f; rsum[r] = 0.f; }
  v8f o[4] = {};
  int half = lane >> 4, col = lane & 15;
  const float scale = 0.125f;  // 64^-0.5

  for (int kt = 0; kt < m0 + 16; kt += 32) {
    v8f s0 = {}, s1 = {};
    {
      v16bf b0 = load_b_bf16_col(Kh, HEADD, kt, 0, lane);
      v16bf b1 = load_b_bf16_col(Kh, HEADD, kt + 16, 0, lane);
      s0 = wmma_bf16(qa0, b0, s0);
      s1 = wmma_bf16(qa0, b1, s1);
      b0 = load_b_bf16_col(Kh, HEADD, kt, 32, lane);
      b1 = load_b_bf16_col(Kh, HEADD, kt + 16, 32, lane);
      s0 = wmma_bf16(qa1, b0, s0);
      s1 = wmma_bf16(qa1, b1, s1);
    }
#pragma unroll
    for (int r = 0; r < 8; ++r) {
      int qrow = m0 + r + 8 * half;
      float x0 = (kt + col <= qrow)      ? s0[r] * scale : -1e30f;
      float x1 = (kt + 16 + col <= qrow) ? s1[r] * scale : -1e30f;
      float mx = fmaxf(x0, x1);
#pragma unroll
      for (int msk = 1; msk < 16; msk <<= 1) mx = fmaxf(mx, __shfl_xor(mx, msk, 32));
      float nm = fmaxf(rmax[r], mx);
      float f = __expf(rmax[r] - nm);
      float p0 = __expf(x0 - nm), p1 = __expf(x1 - nm);
      float ps = p0 + p1;
#pragma unroll
      for (int msk = 1; msk < 16; msk <<= 1) ps += __shfl_xor(ps, msk, 32);
      rsum[r] = rsum[r] * f + ps;
      rmax[r] = nm;
#pragma unroll
      for (int j = 0; j < 4; ++j) o[j][r] *= f;
      int prow = r + 8 * half;
      pbuf[prow * 40 + col] = (bf16_t)p0;
      pbuf[prow * 40 + 16 + col] = (bf16_t)p1;
    }
    asm volatile("s_wait_dscnt 0" ::: "memory");
    v16bf pa = load_a_bf16(pbuf, 40, 0, 0, lane);
#pragma unroll
    for (int j = 0; j < 4; ++j) {
      v16bf bv = load_b_bf16_row(Vh, HEADD, kt, 16 * j, lane);
      o[j] = wmma_bf16(pa, bv, o[j]);
    }
  }
#pragma unroll
  for (int r = 0; r < 8; ++r) {
    float inv = 1.f / rsum[r];
    int trow = m0 + r + 8 * half;
#pragma unroll
    for (int j = 0; j < 4; ++j)
      O[(size_t)trow * (NHEADS * HEADD) + head * HEADD + 16 * j + col] =
          (bf16_t)(o[j][r] * inv);
  }
}

// ---------------------------------------------------------------------------
// Output projection + residual: hid = hidden + attn @ wo^T  (fp32 out)
// ---------------------------------------------------------------------------
__global__ void o_proj_kernel(const bf16_t* __restrict__ A,
                              const float* __restrict__ W,
                              const float* __restrict__ resid,
                              float* __restrict__ out) {
  __shared__ bf16_t atile[2][64 * ATILE_STRIDE];
  int tid = threadIdx.x;
  int lane = tid & 31, wid = tid >> 5;
  int row = tid >> 1, seg = tid & 1;
  int mblk = blockIdx.x * 64;
  int m0 = mblk + wid * 16;
  int n0 = blockIdx.y * 64;
  const bf16_t* grow = A + (size_t)(mblk + row) * HDIM + seg * 16;
  unsigned lrow0 = lds_off_of(&atile[0][0]) + (unsigned)(row * 80 + seg * 32);
  unsigned lrow1 = lds_off_of(&atile[1][0]) + (unsigned)(row * 80 + seg * 32);

  v8f acc[4] = {};
  async_stage_rowpair(lrow0, grow);
  for (int k = 0; k < HDIM; k += 32) {
    int cur = (k >> 5) & 1;
    if (k + 32 < HDIM) {
      async_stage_rowpair(cur ? lrow0 : lrow1, grow + (k + 32));
      S_WAIT_ASYNCCNT(2);
    } else {
      S_WAIT_ASYNCCNT(0);
    }
    v16bf a = load_a_bf16(atile[cur], ATILE_STRIDE, wid * 16, 0, lane);
#pragma unroll
    for (int j = 0; j < 4; ++j)
      acc[j] = wmma_bf16(a, load_b_w32(W, HDIM, n0 + 16 * j, k, lane), acc[j]);
  }
  int half = lane >> 4, col = lane & 15;
#pragma unroll
  for (int r = 0; r < 8; ++r) {
#pragma unroll
    for (int j = 0; j < 4; ++j) {
      size_t idx = (size_t)(m0 + r + 8 * half) * HDIM + n0 + 16 * j + col;
      out[idx] = resid[idx] + acc[j][r];
    }
  }
}

// ---------------------------------------------------------------------------
// Router: logits, softmax, top-2, renormalize. One wave per token.
// ---------------------------------------------------------------------------
__global__ void router_kernel(const bf16_t* __restrict__ x2,
                              const float* __restrict__ rw,
                              float* __restrict__ gate) {
  int lane = threadIdx.x & 31, wid = threadIdx.x >> 5;
  int t = blockIdx.x * 8 + wid;
  float lg[NEXP];
#pragma unroll
  for (int e = 0; e < NEXP; ++e) lg[e] = 0.f;
  for (int i = lane; i < HDIM; i += 32) {
    float xv = (float)x2[(size_t)t * HDIM + i];
#pragma unroll
    for (int e = 0; e < NEXP; ++e) lg[e] += xv * rw[e * HDIM + i];
  }
#pragma unroll
  for (int e = 0; e < NEXP; ++e) {
#pragma unroll
    for (int m = 16; m >= 1; m >>= 1) lg[e] += __shfl_xor(lg[e], m, 32);
  }
  if (lane == 0) {
    float mx = lg[0];
#pragma unroll
    for (int e = 1; e < NEXP; ++e) mx = fmaxf(mx, lg[e]);
    float p[NEXP];
#pragma unroll
    for (int e = 0; e < NEXP; ++e) p[e] = __expf(lg[e] - mx);
    int i0 = 0;
#pragma unroll
    for (int e = 1; e < NEXP; ++e)
      if (p[e] > p[i0]) i0 = e;
    int i1 = (i0 == 0) ? 1 : 0;
#pragma unroll
    for (int e = 0; e < NEXP; ++e)
      if (e != i0 && p[e] > p[i1]) i1 = e;
    float tt = p[i0] + p[i1];
#pragma unroll
    for (int e = 0; e < NEXP; ++e) gate[(size_t)t * NEXP + e] = 0.f;
    gate[(size_t)t * NEXP + i0] = p[i0] / tt;
    gate[(size_t)t * NEXP + i1] = p[i1] / tt;
  }
}

// ---------------------------------------------------------------------------
// FFN stage 1 (per expert): hbuf = silu(x2 @ w1^T) * (x2 @ w3^T)  (bf16)
// Whole 64x1024 bf16 A-tile staged by the Tensor Data Mover (one descriptor,
// LDS padding 16B per 1024B for bank spread), tracked by TENSORcnt.
// ---------------------------------------------------------------------------
__global__ void ffn1_kernel(const bf16_t* __restrict__ x2,
                            const float* __restrict__ w1e,
                            const float* __restrict__ w3e,
                            bf16_t* __restrict__ hbuf) {
  __shared__ bf16_t atile[(64 * TDM_ROW_BYTES) / 2 + 16];
  int tid = threadIdx.x;
  int lane = tid & 31, wid = tid >> 5;
  int mblk = blockIdx.x * 64;
  int m0 = mblk + wid * 16;
  int n0 = blockIdx.y * 64;

  if (wid == 0) {
    unsigned lds = lds_off_of(&atile[0]);
    unsigned long long ga =
        (unsigned long long)(size_t)(x2 + (size_t)mblk * HDIM);
    v4u g0;
    g0.x = 1u;                                     // count=1 (valid, user mode)
    g0.y = lds;                                    // lds_addr (bytes)
    g0.z = (unsigned)(ga & 0xffffffffu);           // global_addr[31:0]
    g0.w = (unsigned)((ga >> 32) & 0x01ffffffu) |  // global_addr[56:32]
           (2u << 30);                             // type=2 (image)
    v8u g1;
    g1[0] = (1u << 16)    // data_size = 2 bytes
          | (1u << 20)    // pad_enable
          | (7u << 22)    // pad_interval = 256 DWORDs (1024B)
          | (3u << 25);   // pad_amount = 4 DWORDs (16B)
    g1[1] = ((unsigned)HDIM & 0xffffu) << 16;      // tensor_dim0[15:0] = 1024
    g1[2] = (((unsigned)HDIM >> 16) & 0xffffu)     // tensor_dim0[31:16]
          | (64u << 16);                           // tensor_dim1[15:0] = 64
    g1[3] = ((unsigned)HDIM << 16);                // tensor_dim1 hi=0, tile_dim0=1024
    g1[4] = 64u;                                   // tile_dim1=64, tile_dim2=0
    g1[5] = (unsigned)HDIM;                        // tensor_dim0_stride lo = 1024
    g1[6] = 0u;                                    // stride hi / dim1_stride lo
    g1[7] = 0u;                                    // dim1_stride hi
    asm volatile("tensor_load_to_lds %0, %1" :: "s"(g0), "s"(g1) : "memory");
    wait_tensorcnt0();
  }
  __syncthreads();

  v8f c1[4] = {}, c3[4] = {};
  for (int k = 0; k < HDIM; k += 32) {
    v16bf a = load_a_tdm(atile, wid * 16, k, lane);
#pragma unroll
    for (int j = 0; j < 4; ++j) {
      c1[j] = wmma_bf16(a, load_b_w32(w1e, HDIM, n0 + 16 * j, k, lane), c1[j]);
      c3[j] = wmma_bf16(a, load_b_w32(w3e, HDIM, n0 + 16 * j, k, lane), c3[j]);
    }
  }
  int half = lane >> 4, col = lane & 15;
#pragma unroll
  for (int r = 0; r < 8; ++r) {
#pragma unroll
    for (int j = 0; j < 4; ++j) {
      float g = c1[j][r];
      float h = (g / (1.f + __expf(-g))) * c3[j][r];
      hbuf[(size_t)(m0 + r + 8 * half) * IDIM + n0 + 16 * j + col] = (bf16_t)h;
    }
  }
}

// ---------------------------------------------------------------------------
// FFN stage 2 (per expert): out = base + gate[t,e] * (hbuf @ w2^T)
// expert==0 reads hid (fuses final residual), else accumulates into out.
// ---------------------------------------------------------------------------
__global__ void ffn2_kernel(const bf16_t* __restrict__ hbuf,
                            const float* __restrict__ w2e,
                            const float* __restrict__ gate,
                            const float* __restrict__ hid,
                            float* __restrict__ out, int expert) {
  __shared__ bf16_t atile[2][64 * ATILE_STRIDE];
  int tid = threadIdx.x;
  int lane = tid & 31, wid = tid >> 5;
  int row = tid >> 1, seg = tid & 1;
  int mblk = blockIdx.x * 64;
  int m0 = mblk + wid * 16;
  int n0 = blockIdx.y * 64;
  const bf16_t* grow = hbuf + (size_t)(mblk + row) * IDIM + seg * 16;
  unsigned lrow0 = lds_off_of(&atile[0][0]) + (unsigned)(row * 80 + seg * 32);
  unsigned lrow1 = lds_off_of(&atile[1][0]) + (unsigned)(row * 80 + seg * 32);

  v8f acc[4] = {};
  async_stage_rowpair(lrow0, grow);
  for (int k = 0; k < IDIM; k += 32) {
    int cur = (k >> 5) & 1;
    if (k + 32 < IDIM) {
      async_stage_rowpair(cur ? lrow0 : lrow1, grow + (k + 32));
      S_WAIT_ASYNCCNT(2);
    } else {
      S_WAIT_ASYNCCNT(0);
    }
    v16bf a = load_a_bf16(atile[cur], ATILE_STRIDE, wid * 16, 0, lane);
#pragma unroll
    for (int j = 0; j < 4; ++j)
      acc[j] = wmma_bf16(a, load_b_w32(w2e, IDIM, n0 + 16 * j, k, lane), acc[j]);
  }
  int half = lane >> 4, col = lane & 15;
#pragma unroll
  for (int r = 0; r < 8; ++r) {
    int m = m0 + r + 8 * half;
    float g = gate[(size_t)m * NEXP + expert];
#pragma unroll
    for (int j = 0; j < 4; ++j) {
      size_t idx = (size_t)m * HDIM + n0 + 16 * j + col;
      float base = (expert == 0) ? hid[idx] : out[idx];
      out[idx] = base + g * acc[j][r];
    }
  }
}

// ---------------------------------------------------------------------------
// Host launch
// ---------------------------------------------------------------------------
extern "C" void kernel_launch(void* const* d_in, const int* in_sizes, int n_in,
                              void* d_out, int out_size, void* d_ws, size_t ws_size,
                              hipStream_t stream) {
  const float* hidden = (const float*)d_in[0];
  const float* cosb   = (const float*)d_in[1];
  const float* sinb   = (const float*)d_in[2];
  const float* n1w    = (const float*)d_in[3];
  const float* n2w    = (const float*)d_in[4];
  const float* wq     = (const float*)d_in[5];
  const float* wk     = (const float*)d_in[6];
  const float* wv     = (const float*)d_in[7];
  const float* wo     = (const float*)d_in[8];
  const float* rw     = (const float*)d_in[9];
  const float* w1     = (const float*)d_in[10];
  const float* w2     = (const float*)d_in[11];
  const float* w3     = (const float*)d_in[12];
  float* out = (float*)d_out;

  char* p = (char*)d_ws;
  auto alloc = [&](size_t bytes) -> char* {
    char* r = p;
    p += (bytes + 255) & ~(size_t)255;
    return r;
  };
  bf16_t* xb   = (bf16_t*)alloc((size_t)S_LEN * HDIM * 2);
  bf16_t* qb   = (bf16_t*)alloc((size_t)NHEADS * S_LEN * HEADD * 2);
  bf16_t* kb   = (bf16_t*)alloc((size_t)KVHEADS * S_LEN * HEADD * 2);
  bf16_t* vb   = (bf16_t*)alloc((size_t)KVHEADS * S_LEN * HEADD * 2);
  bf16_t* ab   = (bf16_t*)alloc((size_t)S_LEN * HDIM * 2);
  float*  hid  = (float*)alloc((size_t)S_LEN * HDIM * 4);
  bf16_t* x2b  = (bf16_t*)alloc((size_t)S_LEN * HDIM * 2);
  float*  gate = (float*)alloc((size_t)S_LEN * NEXP * 4);
  bf16_t* hbuf = (bf16_t*)alloc((size_t)S_LEN * IDIM * 2);

  rmsnorm_kernel<<<S_LEN, 256, 0, stream>>>(hidden, n1w, xb);
  qkv_rope_kernel<<<dim3(S_LEN / 64, 24), 128, 0, stream>>>(xb, wq, wk, wv, cosb,
                                                            sinb, qb, kb, vb);
  attn_kernel<<<dim3(S_LEN / 16, NHEADS), 32, 0, stream>>>(qb, kb, vb, ab);
  o_proj_kernel<<<dim3(S_LEN / 64, HDIM / 64), 128, 0, stream>>>(ab, wo, hidden, hid);
  rmsnorm_kernel<<<S_LEN, 256, 0, stream>>>(hid, n2w, x2b);
  router_kernel<<<S_LEN / 8, 256, 0, stream>>>(x2b, rw, gate);
  for (int e = 0; e < NEXP; ++e) {
    ffn1_kernel<<<dim3(S_LEN / 64, IDIM / 64), 128, 0, stream>>>(
        x2b, w1 + (size_t)e * IDIM * HDIM, w3 + (size_t)e * IDIM * HDIM, hbuf);
    ffn2_kernel<<<dim3(S_LEN / 64, HDIM / 64), 128, 0, stream>>>(
        hbuf, w2 + (size_t)e * HDIM * IDIM, gate, hid, out, e);
  }
  (void)in_sizes; (void)n_in; (void)out_size; (void)ws_size;
}
